// ESSPFormerModel_47459388620928
// MI455X (gfx1250) — matrix operand order repaired and checked by reading, not compiled, not run on hardware
//
#include <hip/hip_runtime.h>
#include <hip/hip_bf16.h>

typedef _Float16 v16h  __attribute__((ext_vector_type(16)));
typedef _Float16 half8 __attribute__((ext_vector_type(8)));
typedef float    v8f   __attribute__((ext_vector_type(8)));
typedef float    f4    __attribute__((ext_vector_type(4)));

#define DMODEL   800
#define NCH      200
#define SEQ      25
#define HDIM     32
#define NCLS     16
#define ROWS     16
#define NTHREADS 256

union V16H { v16h v; half8 h[2]; };
union V8F  { v8f v; float f[8]; };

__device__ __forceinline__ float gelu_exact(float t) {
  return 0.5f * t * (1.0f + erff(t * 0.70710678118654752f));
}

// ---------------------------------------------------------------------------
// Prep: convert w1 (E,800,32) and w2 (E,32,800) f32 -> f16, N-major layouts:
//   w1t[e][n][k] (n<32,k<800), w2t[e][n][k] (n<800,k<32)
// so every WMMA B-fragment is two contiguous 16B loads per lane.
// ---------------------------------------------------------------------------
__global__ __launch_bounds__(NTHREADS) void prep_weights_f16(
    const float* __restrict__ w1, const float* __restrict__ w2,
    _Float16* __restrict__ w1t, _Float16* __restrict__ w2t) {
  int idx = blockIdx.x * blockDim.x + threadIdx.x;
  const int per_e = DMODEL * HDIM;
  if (idx >= 2 * per_e) return;
  int e = idx / per_e;
  int rem = idx - e * per_e;
  {
    int k = rem / HDIM;           // 0..799
    int n = rem - k * HDIM;       // 0..31
    w1t[e * per_e + n * DMODEL + k] = (_Float16)w1[idx];
  }
  {
    int k = rem / DMODEL;         // 0..31
    int n = rem - k * DMODEL;     // 0..799
    w2t[e * per_e + n * HDIM + k] = (_Float16)w2[e * per_e + k * DMODEL + n];
  }
}

// ---------------------------------------------------------------------------
// LayerNorm of hbuf rows -> f16 ybuf. 16 threads cooperate per row.
// ---------------------------------------------------------------------------
__device__ __forceinline__ void layernorm_f16(
    float (&hbuf)[ROWS][DMODEL], _Float16 (&ybuf)[ROWS][DMODEL],
    const float* __restrict__ g, const float* __restrict__ b,
    float (&red0)[NTHREADS], float (&red1)[NTHREADS], float (&stats)[ROWS][2],
    int tid) {
  int r = tid >> 4, q = tid & 15;
  float s = 0.f, ss = 0.f;
  for (int i = q; i < DMODEL; i += 16) { float v = hbuf[r][i]; s += v; ss += v * v; }
  red0[tid] = s; red1[tid] = ss;
  __syncthreads();
  if (q == 0) {
    float S = 0.f, SS = 0.f;
#pragma unroll
    for (int j = 0; j < 16; ++j) { S += red0[(r << 4) + j]; SS += red1[(r << 4) + j]; }
    float m   = S * (1.f / DMODEL);
    float var = SS * (1.f / DMODEL) - m * m;
    stats[r][0] = m;
    stats[r][1] = rsqrtf(fmaxf(var, 0.f) + 1e-5f);
  }
  __syncthreads();
  float m = stats[r][0], rs = stats[r][1];
  for (int i = q; i < DMODEL; i += 16) {
    float yv = (hbuf[r][i] - m) * rs * g[i] + b[i];
    ybuf[r][i] = (_Float16)yv;
  }
  __syncthreads();
}

// ---------------------------------------------------------------------------
// Fused model kernel: one workgroup = one 16-row WMMA M-tile.
// ---------------------------------------------------------------------------
__global__ __launch_bounds__(NTHREADS) void essp_fused(
    const float* __restrict__ x,
    const float* __restrict__ lnAg, const float* __restrict__ lnAb,
    const float* __restrict__ lnBg, const float* __restrict__ lnBb,
    const _Float16* __restrict__ w1t, const float* __restrict__ b1,
    const _Float16* __restrict__ w2t, const float* __restrict__ b2,
    const float* __restrict__ fcw, const float* __restrict__ fcb,
    float* __restrict__ out, int B) {
  __shared__ __align__(16) float    hbuf[ROWS][DMODEL];   // residual stream (f32)
  __shared__ __align__(16) _Float16 ybuf[ROWS][DMODEL];   // LN outputs (f16)
  __shared__ __align__(16) _Float16 act[ROWS][HDIM];      // gelu(z@w1+b1) (f16)
  __shared__ float red0[NTHREADS], red1[NTHREADS];
  __shared__ float stats[ROWS][2];

  const int tid   = threadIdx.x;
  const int wg    = blockIdx.x;
  const int lane  = tid & 31;
  const int wv    = tid >> 5;
  const int half_ = lane >> 4;   // 0 or 1 (half-wave)
  const int mrow  = lane & 15;   // M (or N) index within 16

  // ---- Stage 1: stream x once (NT, 128-bit, fully coalesced) and build
  //      h[4c..4c+3] = [min, mean, last, max] per channel. Each task owns
  //      4 consecutive channels -> one b128 per timestep per lane.
  for (int idx = tid; idx < ROWS * (NCH / 4); idx += NTHREADS) {
    int r = idx / (NCH / 4), c4 = idx - r * (NCH / 4);
    int row = wg * ROWS + r; if (row >= B) row = B - 1;
    const f4* xp = (const f4*)(x + (size_t)row * (SEQ * NCH)) + c4;
    f4 v0 = __builtin_nontemporal_load(xp);
    f4 mn = v0, mx = v0, sm = v0;
#pragma unroll
    for (int s = 1; s < SEQ - 1; ++s) {
      f4 v = __builtin_nontemporal_load(xp + s * (NCH / 4));
      mn = __builtin_elementwise_min(mn, v);
      mx = __builtin_elementwise_max(mx, v);
      sm = sm + v;
    }
    f4 last = __builtin_nontemporal_load(xp + (SEQ - 1) * (NCH / 4));
#pragma unroll
    for (int j = 0; j < 4; ++j) {
      int c = 4 * c4 + j;
      hbuf[r][4 * c + 0] = mn[j];
      hbuf[r][4 * c + 1] = sm[j] * (1.f / (SEQ - 1));
      hbuf[r][4 * c + 2] = last[j];
      hbuf[r][4 * c + 3] = mx[j];
    }
  }
  __syncthreads();

  // ---- Stage 2: E = 2 blocks ----
  for (int e = 0; e < 2; ++e) {
    // LayerNorm A -> ybuf
    layernorm_f16(hbuf, ybuf, lnAg + e * DMODEL, lnAb + e * DMODEL,
                  red0, red1, stats, tid);

    // sppcf(y) added into h
    for (int idx = tid; idx < ROWS * DMODEL; idx += NTHREADS) {
      int r = idx / DMODEL, j = idx - r * DMODEL;
      int k = j >> 2, mm = j & 3;
      float val;
      if (mm == 0) {
        val = 0.5f * ((float)ybuf[r][4 * k] + (float)ybuf[r][4 * k + 1]);
      } else if (mm == 1) {
        val = 0.25f * ((float)ybuf[r][4 * k] + (float)ybuf[r][4 * k + 1] +
                       (float)ybuf[r][4 * k + 2] + (float)ybuf[r][4 * k + 3]);
      } else if (mm == 2) {
        val = 0.5f * ((float)ybuf[r][4 * k + 2] + (float)ybuf[r][4 * k + 3]);
      } else {
        int kk = (k < NCH - 3) ? k : (k - 4);   // f2 tail replication
        float ssum = 0.f;
#pragma unroll
        for (int t = 0; t < 16; ++t) ssum += (float)ybuf[r][4 * kk + t];
        val = ssum * (1.f / 16.f);
      }
      hbuf[r][j] += val;
    }
    __syncthreads();

    // LayerNorm B -> ybuf (MLP input z, f16)
    layernorm_f16(hbuf, ybuf, lnBg + e * DMODEL, lnBb + e * DMODEL,
                  red0, red1, stats, tid);

    // matmul1: t1 = z(16x800) @ w1(800x32), waves 0..1 each own one 16-col tile
    if (wv < 2) {
      int n = wv * 16 + mrow;
      const _Float16* wrow = w1t + e * (DMODEL * HDIM) + n * DMODEL;
      V8F cacc;
#pragma unroll
      for (int i = 0; i < 8; ++i) cacc.f[i] = 0.f;
      for (int ks = 0; ks < DMODEL / 32; ++ks) {
        int kb = ks * 32;
        V16H a, bb;
        // A (16x32 f16): per lane K = {8h..8h+7} ∪ {16+8h..16+8h+7}
        a.h[0] = *(const half8*)&ybuf[mrow][kb + 8 * half_];
        a.h[1] = *(const half8*)&ybuf[mrow][kb + 16 + 8 * half_];
        // B (32x16 f16): per lane K = {8h..8h+7} ∪ {16+8h..16+8h+7}
        bb.h[0] = *(const half8*)&wrow[kb + 8 * half_];
        bb.h[1] = *(const half8*)&wrow[kb + 16 + 8 * half_];
        cacc.v = __builtin_amdgcn_wmma_f32_16x16x32_f16(
            false, a.v, false, bb.v, (short)0, cacc.v, false, false);
      }
      float bias = b1[e * HDIM + n];
#pragma unroll
      for (int v = 0; v < 8; ++v) {     // C/D: row = vgpr + 8*half, col = lane%16
        act[v + 8 * half_][n] = (_Float16)gelu_exact(cacc.f[v] + bias);
      }
    }
    __syncthreads();

    // matmul2: z2 = act(16x32) @ w2(32x800); 50 N-tiles over 8 waves; h += z2+b2
    {
      V16H a;
      a.h[0] = *(const half8*)&act[mrow][8 * half_];
      a.h[1] = *(const half8*)&act[mrow][16 + 8 * half_];
      for (int tile = wv; tile < DMODEL / 16; tile += 8) {   // wave-uniform loop
        int n = tile * 16 + mrow;
        const _Float16* wrow = w2t + e * (DMODEL * HDIM) + n * HDIM;
        V16H bb;
        bb.h[0] = *(const half8*)&wrow[8 * half_];
        bb.h[1] = *(const half8*)&wrow[16 + 8 * half_];
        V8F cacc;
#pragma unroll
        for (int i = 0; i < 8; ++i) cacc.f[i] = 0.f;
        cacc.v = __builtin_amdgcn_wmma_f32_16x16x32_f16(
            false, a.v, false, bb.v, (short)0, cacc.v, false, false);
        float bias = b2[e * DMODEL + n];
#pragma unroll
        for (int v = 0; v < 8; ++v) hbuf[v + 8 * half_][n] += cacc.f[v] + bias;
      }
    }
    __syncthreads();
  }

  // ---- Stage 3: pool over groups of 4 + classifier (16x200 @ 200x16) ----
  {
    int r = tid >> 4, n = tid & 15;
    float acc = fcb[n];
    for (int c = 0; c < NCH; ++c) {
      float p = 0.25f * (hbuf[r][4 * c] + hbuf[r][4 * c + 1] +
                         hbuf[r][4 * c + 2] + hbuf[r][4 * c + 3]);
      acc += p * fcw[c * NCLS + n];
    }
    int row = wg * ROWS + r;
    if (row < B) out[row * NCLS + n] = acc;
  }
}

// ---------------------------------------------------------------------------
extern "C" void kernel_launch(void* const* d_in, const int* in_sizes, int n_in,
                              void* d_out, int out_size, void* d_ws, size_t ws_size,
                              hipStream_t stream) {
  const float* x    = (const float*)d_in[0];
  const float* lnAg = (const float*)d_in[1];
  const float* lnAb = (const float*)d_in[2];
  const float* lnBg = (const float*)d_in[3];
  const float* lnBb = (const float*)d_in[4];
  const float* w1   = (const float*)d_in[5];
  const float* b1   = (const float*)d_in[6];
  const float* w2   = (const float*)d_in[7];
  const float* b2   = (const float*)d_in[8];
  const float* fcw  = (const float*)d_in[9];
  const float* fcb  = (const float*)d_in[10];

  int B = in_sizes[0] / (SEQ * NCH);

  _Float16* w1t = (_Float16*)d_ws;                 // 2*800*32 halves = 102400 B
  _Float16* w2t = w1t + 2 * DMODEL * HDIM;         // +102400 B  (ws >= 200 KB)

  int total = 2 * DMODEL * HDIM;
  prep_weights_f16<<<(total + NTHREADS - 1) / NTHREADS, NTHREADS, 0, stream>>>(
      w1, w2, w1t, w2t);

  essp_fused<<<(B + ROWS - 1) / ROWS, NTHREADS, 0, stream>>>(
      x, lnAg, lnAb, lnBg, lnBb, w1t, b1, w2t, b2, fcw, fcb, (float*)d_out, B);
}
